// EPMoeBlock_38920993636978
// MI455X (gfx1250) — matrix operand order repaired and checked by reading, not compiled, not run on hardware
//
#include <hip/hip_runtime.h>
#include <hip/hip_bf16.h>

#define T_TOK 4096
#define HID   1024
#define INTER 2048
#define NEXP  8
#define BM    128
#define BN    128

typedef __attribute__((ext_vector_type(16))) __bf16 v16bf;
typedef __attribute__((ext_vector_type(8)))  float  v8f;

__device__ __forceinline__ v8f wmma_bf16(v16bf a, v16bf b, v8f c) {
  // (neg_a, A, neg_b, B, c_mod, C, reuse_a, reuse_b)
  return __builtin_amdgcn_wmma_f32_16x16x32_bf16(false, a, false, b, (short)0, c, false, false);
}

// ---- A-fragment index helpers (16-bit A 16x32, contiguous k order matching B) ----
// lane = a_hi(kf)*16 + m ; element j = a_j(kf)
__device__ __forceinline__ int a_hi(int kf) { return kf >> 4; }
__device__ __forceinline__ int a_j (int kf) { return kf & 15; }
// ---- B-fragment: lane = (kf>>4)*16 + n ; element j = kf & 15 ----

// Async DMA: global -> LDS, one dword per lane (ASYNCcnt tracked).
__device__ __forceinline__ void async_copy_b32(void* lds_ptr, const void* gptr) {
  uint32_t lds_off = (uint32_t)(uintptr_t)lds_ptr;   // low 32 bits of LDS aperture = DS offset
  asm volatile("global_load_async_to_lds_b32 %0, %1, off"
               :: "v"(lds_off), "v"((uint64_t)(uintptr_t)gptr)
               : "memory");
}
__device__ __forceinline__ void wait_async() {
  asm volatile("s_wait_asynccnt 0x0" ::: "memory");
}

__global__ __launch_bounds__(256) void zero_out_kernel(float* __restrict__ out) {
  int idx = (blockIdx.x * 256 + threadIdx.x) * 4;
  float4 z = {0.f, 0.f, 0.f, 0.f};
  *(float4*)(out + idx) = z;
}

__global__ __launch_bounds__(256)
void router_kernel(const float* __restrict__ x, const float* __restrict__ gate_w,
                   float* __restrict__ combine, float* __restrict__ logits_out) {
  __shared__ float sgw[NEXP * HID]; // 32 KB
  const int tid = threadIdx.x;
  for (int i = tid; i < NEXP * HID; i += 256) sgw[i] = gate_w[i];
  __syncthreads();
  const int wave = tid >> 5, lane = tid & 31;
  const int t = blockIdx.x * 8 + wave;
  float acc[NEXP];
#pragma unroll
  for (int e = 0; e < NEXP; ++e) acc[e] = 0.f;
  const float* xr = x + (size_t)t * HID;
  for (int k = lane; k < HID; k += 32) {
    float xv = xr[k];
#pragma unroll
    for (int e = 0; e < NEXP; ++e) acc[e] += xv * sgw[e * HID + k];
  }
#pragma unroll
  for (int e = 0; e < NEXP; ++e)
    for (int off = 16; off > 0; off >>= 1) acc[e] += __shfl_xor(acc[e], off, 32);
  if (lane == 0) {
    float mx = acc[0];
    for (int e = 1; e < NEXP; ++e) mx = fmaxf(mx, acc[e]);
    float p[NEXP], s = 0.f;
    for (int e = 0; e < NEXP; ++e) { p[e] = __expf(acc[e] - mx); s += p[e]; }
    float inv = 1.f / s;
    for (int e = 0; e < NEXP; ++e) p[e] *= inv;
    int i1 = 0;
    for (int e = 1; e < NEXP; ++e) if (p[e] > p[i1]) i1 = e;
    int i2 = (i1 == 0) ? 1 : 0;
    for (int e = 0; e < NEXP; ++e) if (e != i1 && p[e] > p[i2]) i2 = e;
    float norm = 1.f / (p[i1] + p[i2]);
    for (int e = 0; e < NEXP; ++e) {
      float w = (e == i1) ? p[i1] * norm : ((e == i2) ? p[i2] * norm : 0.f);
      combine[t * NEXP + e]    = w;
      logits_out[t * NEXP + e] = acc[e];
    }
  }
}

// h_w[t, i] = silu(x@Wg) * (x@Wu) * combine[t, e]   (bf16, pre-weighted)
__global__ __launch_bounds__(256)
void gateup_kernel(const float* __restrict__ x, const float* __restrict__ w_gate,
                   const float* __restrict__ w_up, const float* __restrict__ combine,
                   __bf16* __restrict__ h_w, int e) {
  __shared__ alignas(32) __bf16 sA [8][32][16]; // 128x32 x tile  (A-frag layout)
  __shared__ alignas(32) __bf16 sBg[8][32][16]; // 32x128 Wg      (B-frag layout)
  __shared__ alignas(32) __bf16 sBu[8][32][16]; // 32x128 Wu
  const int tid = threadIdx.x, lane = tid & 31, wave = tid >> 5;
  const int mp = (wave & 3) * 2;   // two 16-row subtiles: mp, mp+1
  const int nh = wave >> 2;        // four 16-col subtiles: nh*4 .. +3
  const int n0 = blockIdx.x * BN, t0 = blockIdx.y * BM;
  const float* Wg = w_gate + (size_t)e * HID * INTER;
  const float* Wu = w_up   + (size_t)e * HID * INTER;

  v8f accg[2][4], accu[2][4];
#pragma unroll
  for (int s = 0; s < 2; ++s)
#pragma unroll
    for (int t = 0; t < 4; ++t) { accg[s][t] = {}; accu[s][t] = {}; }

  for (int k0 = 0; k0 < HID; k0 += 32) {
#pragma unroll
    for (int i = 0; i < 16; ++i) {           // stage x: 128 rows x 32 k, cvt f32->bf16
      int idx = tid + i * 256;
      int m = idx >> 5, kf = idx & 31;
      float v = x[(size_t)(t0 + m) * HID + k0 + kf];
      sA[m >> 4][a_hi(kf) * 16 + (m & 15)][a_j(kf)] = (__bf16)v;
    }
#pragma unroll
    for (int i = 0; i < 16; ++i) {           // stage Wg/Wu: 32 k x 128 n
      int idx = tid + i * 256;
      int kf = idx >> 7, n = idx & 127;
      int nt = n >> 4, bl = (kf >> 4) * 16 + (n & 15), bj = kf & 15;
      size_t ga = (size_t)(k0 + kf) * INTER + n0 + n;
      sBg[nt][bl][bj] = (__bf16)Wg[ga];
      sBu[nt][bl][bj] = (__bf16)Wu[ga];
    }
    if (k0 + 32 < HID) {                     // prefetch next k-chunk into L2/WGP$
      int kf = tid >> 7, n = tid & 127;
      size_t gp = (size_t)(k0 + 32 + kf) * INTER + n0 + n;
      __builtin_prefetch(Wg + gp, 0, 1);
      __builtin_prefetch(Wu + gp, 0, 1);
    }
    __syncthreads();
    v16bf a0 = *(const v16bf*)(&sA[mp    ][lane][0]);
    v16bf a1 = *(const v16bf*)(&sA[mp + 1][lane][0]);
#pragma unroll
    for (int t = 0; t < 4; ++t) {
      int nt = nh * 4 + t;
      v16bf bg = *(const v16bf*)(&sBg[nt][lane][0]);
      accg[0][t] = wmma_bf16(a0, bg, accg[0][t]);
      accg[1][t] = wmma_bf16(a1, bg, accg[1][t]);
      v16bf bu = *(const v16bf*)(&sBu[nt][lane][0]);
      accu[0][t] = wmma_bf16(a0, bu, accu[0][t]);
      accu[1][t] = wmma_bf16(a1, bu, accu[1][t]);
    }
    __syncthreads();
  }

  const int hi = lane >> 4, nloc = lane & 15;
#pragma unroll
  for (int s = 0; s < 2; ++s) {
    float cw[8];
#pragma unroll
    for (int v = 0; v < 8; ++v)
      cw[v] = combine[(t0 + (mp + s) * 16 + hi * 8 + v) * NEXP + e];
#pragma unroll
    for (int t = 0; t < 4; ++t) {
      int ncol = n0 + (nh * 4 + t) * 16 + nloc;
#pragma unroll
      for (int v = 0; v < 8; ++v) {
        int tok = t0 + (mp + s) * 16 + hi * 8 + v;  // D layout: M = vgpr + 8*(lane>>4)
        float g = accg[s][t][v], u = accu[s][t][v];
        float sg = g / (1.f + __expf(-g));          // silu
        h_w[(size_t)tok * INTER + ncol] = (__bf16)(sg * u * cw[v]);
      }
    }
  }
}

// out[t, h] += h_w @ w_down[e]
__global__ __launch_bounds__(256)
void down_kernel(const __bf16* __restrict__ h_w, const float* __restrict__ w_down,
                 float* __restrict__ out, int e) {
  __shared__ alignas(32) __bf16 sA[8][32][16]; // 128x32 h tile (async DMA'd)
  __shared__ alignas(32) __bf16 sB[8][32][16]; // 32x128 Wd
  const int tid = threadIdx.x, lane = tid & 31, wave = tid >> 5;
  const int mp = (wave & 3) * 2;
  const int nh = wave >> 2;
  const int n0 = blockIdx.x * BN, t0 = blockIdx.y * BM;
  const float* Wd = w_down + (size_t)e * INTER * HID;

  v8f acc[2][4];
#pragma unroll
  for (int s = 0; s < 2; ++s)
#pragma unroll
    for (int t = 0; t < 4; ++t) acc[s][t] = {};

  for (int k0 = 0; k0 < INTER; k0 += 32) {
#pragma unroll
    for (int i = 0; i < 8; ++i) {            // async stage h (bf16): dword pairs, ASYNCcnt
      int pid = tid + i * 256;               // 2048 pairs
      int m = pid >> 4, kf = (pid & 15) * 2; // (kf,kf+1) -> (j,j+1), same lane
      async_copy_b32(&sA[m >> 4][a_hi(kf) * 16 + (m & 15)][a_j(kf)],
                     h_w + (size_t)(t0 + m) * INTER + k0 + kf);
    }
#pragma unroll
    for (int i = 0; i < 16; ++i) {           // stage Wd: 32 k x 128 n, cvt
      int idx = tid + i * 256;
      int kf = idx >> 7, n = idx & 127;
      sB[n >> 4][(kf >> 4) * 16 + (n & 15)][kf & 15] =
          (__bf16)Wd[(size_t)(k0 + kf) * HID + n0 + n];
    }
    if (k0 + 32 < INTER) {
      int kf = tid >> 7, n = tid & 127;
      __builtin_prefetch(Wd + (size_t)(k0 + 32 + kf) * HID + n0 + n, 0, 1);
    }
    wait_async();                            // ASYNCcnt==0 before workgroup barrier
    __syncthreads();
    v16bf a0 = *(const v16bf*)(&sA[mp    ][lane][0]);
    v16bf a1 = *(const v16bf*)(&sA[mp + 1][lane][0]);
#pragma unroll
    for (int t = 0; t < 4; ++t) {
      v16bf b = *(const v16bf*)(&sB[nh * 4 + t][lane][0]);
      acc[0][t] = wmma_bf16(a0, b, acc[0][t]);
      acc[1][t] = wmma_bf16(a1, b, acc[1][t]);
    }
    __syncthreads();
  }

  const int hi = lane >> 4, nloc = lane & 15;
#pragma unroll
  for (int s = 0; s < 2; ++s)
#pragma unroll
    for (int t = 0; t < 4; ++t) {
      int ncol = n0 + (nh * 4 + t) * 16 + nloc;
#pragma unroll
      for (int v = 0; v < 8; ++v) {
        size_t oa = (size_t)(t0 + (mp + s) * 16 + hi * 8 + v) * HID + ncol;
        out[oa] += acc[s][t][v];             // one owner per element per launch
      }
    }
}

extern "C" void kernel_launch(void* const* d_in, const int* in_sizes, int n_in,
                              void* d_out, int out_size, void* d_ws, size_t ws_size,
                              hipStream_t stream) {
  const float* x      = (const float*)d_in[0];
  const float* gate_w = (const float*)d_in[1];
  const float* w_gate = (const float*)d_in[2];
  const float* w_up   = (const float*)d_in[3];
  const float* w_down = (const float*)d_in[4];
  float* out    = (float*)d_out;                                  // [T][HID]
  float* logits = (float*)d_out + (size_t)T_TOK * HID;            // [T][NEXP]

  float*  combine = (float*)d_ws;                                 // 128 KB
  __bf16* h_w = (__bf16*)((char*)d_ws + (size_t)T_TOK * NEXP * sizeof(float)); // 16 MB

  zero_out_kernel<<<dim3((T_TOK * HID) / 1024), 256, 0, stream>>>(out);
  router_kernel<<<dim3(T_TOK / 8), 256, 0, stream>>>(x, gate_w, combine, logits);
  for (int e = 0; e < NEXP; ++e) {
    gateup_kernel<<<dim3(INTER / BN, T_TOK / BM), 256, 0, stream>>>(x, w_gate, w_up,
                                                                    combine, h_w, e);
    down_kernel<<<dim3(HID / BN, T_TOK / BM), 256, 0, stream>>>(h_w, w_down, out, e);
  }
}
